// Model34_65962107732444
// MI455X (gfx1250) — matrix operand order repaired and checked by reading, not compiled, not run on hardware
//
#include <hip/hip_runtime.h>
#include <hip/hip_bf16.h>
#include <math.h>

#define E    30
#define E4   120
#define NH   3
#define HD   10
#define Tn   512
#define Bn   4096
#define EPSI 1e-5f

typedef __attribute__((ext_vector_type(16))) __bf16 v16bf;
typedef __attribute__((ext_vector_type(8)))  float  v8f;

// Branchless fast transcendentals: v_exp_f32 + v_rcp_f32, no libm branches.
__device__ __forceinline__ float sigmoidf_(float x) {
    return __builtin_amdgcn_rcpf(1.0f + __expf(-x));
}
__device__ __forceinline__ float tanhf_(float x) {
    // tanh(x) = 2/(1+exp(-2x)) - 1 ; saturates to +/-1 at the extremes.
    return 2.0f * __builtin_amdgcn_rcpf(1.0f + __expf(-2.0f * x)) - 1.0f;
}

// ---------------------------------------------------------------------------
// Kernel 1: full LSTM scan. One wave handles 16 batch rows for all T steps.
//
// Gate columns are permuted to n = gate*32 + slot (slot = hidden unit, 30 of
// 32 used) so that WMMA tile nt carries gate nt>>1 for slots (nt&1)*16+lane.
// After the WMMAs each lane holds i,f,g,o for its own (row, slot) pairs in
// registers -> whole cell update is register-resident; c never touches LDS.
//
// Per step, per tile:  D = x_aug @ [w_ih ; bias]^T  +  (h @ w_hh^T)
// where x_aug = [x0..x4, 1, 0...] rides the K dim of a second chained WMMA,
// with the bias in B-fragment row kk==5.
// ---------------------------------------------------------------------------
__global__ __launch_bounds__(32)
void lstm_kernel(const float* __restrict__ input, const float* __restrict__ w_ih,
                 const float* __restrict__ w_hh, const float* __restrict__ b_ih,
                 const float* __restrict__ b_hh,
                 float* __restrict__ hN, float* __restrict__ cN)
{
    __shared__ float x_s[16 * 5];
    __shared__ float h_s[16 * 32];   // K padded to 32, cols 30..31 stay zero

    const int lane = threadIdx.x;
    const int b0   = blockIdx.x * 16;
    const int lg   = lane >> 4;      // lane group (0/1)
    const int ln   = lane & 15;

    for (int e = lane; e < 16 * 32; e += 32) h_s[e] = 0.0f;

    // B fragments (K x N, N = permuted gate col): lane l -> N-slot = l&15,
    // half i -> K = i + 16*(l>>4).
    v16bf wbh[8], wbx[8];
#pragma unroll
    for (int nt = 0; nt < 8; ++nt) {
        const int gate = nt >> 1;
        const int slot = (nt & 1) * 16 + ln;
        const int grow = gate * E + slot;       // row of w_ih / w_hh (if slot<E)
        const bool vld = (slot < E);
#pragma unroll
        for (int i = 0; i < 16; ++i) {
            const int kk = i + 16 * lg;
            float wh = (vld && kk < E) ? w_hh[grow * E + kk] : 0.0f;
            float wx = 0.0f;
            if (vld) {
                if (kk < 5)       wx = w_ih[grow * 5 + kk];
                else if (kk == 5) wx = b_ih[grow] + b_hh[grow];   // bias via x_aug[5]=1
            }
            wbh[nt][i] = (__bf16)wh;
            wbx[nt][i] = (__bf16)wx;
        }
    }

    // Cell state lives in registers: (rows r+8*lg) x (slots ln, ln+16).
    float c_reg[8][2];
#pragma unroll
    for (int r = 0; r < 8; ++r) { c_reg[r][0] = 0.0f; c_reg[r][1] = 0.0f; }

    __syncthreads();

    for (int t = 0; t < Tn; ++t) {
        // Stage this step's 16x5 input slice.
        for (int e = lane; e < 16 * 5; e += 32)
            x_s[e] = input[(size_t)(b0 + e / 5) * (Tn * 5) + t * 5 + (e % 5)];
        if (t + 1 < Tn && lane < 16)
            __builtin_prefetch(&input[(size_t)(b0 + lane) * (Tn * 5) + (t + 1) * 5], 0, 1);
        __syncthreads();

        // A fragments: lane l -> M = l&15, half i -> K = i + 16*(l>>4)
        v16bf ah, ax;
#pragma unroll
        for (int i = 0; i < 16; ++i) {
            const int kk = i + 16 * lg;
            ah[i] = (__bf16)h_s[ln * 32 + kk];
            float xv = 0.0f;
            if (kk < 5)       xv = x_s[ln * 5 + kk];
            else if (kk == 5) xv = 1.0f;
            ax[i] = (__bf16)xv;
        }

        const v8f cz = {};
        // Two slot-halves; only 4 accumulators live at a time.
#pragma unroll
        for (int s = 0; s < 2; ++s) {
            v8f dg[4];
#pragma unroll
            for (int gate = 0; gate < 4; ++gate) {
                const int nt = gate * 2 + s;
                v8f t0 = __builtin_amdgcn_wmma_f32_16x16x32_bf16(
                    false, ah, false, wbh[nt], (short)0, cz, false, false);
                dg[gate] = __builtin_amdgcn_wmma_f32_16x16x32_bf16(
                    false, ax, false, wbx[nt], (short)0, t0, false, false);
            }
            const int slot = ln + 16 * s;
#pragma unroll
            for (int r = 0; r < 8; ++r) {
                const float ig = sigmoidf_(dg[0][r]);
                const float fg = sigmoidf_(dg[1][r]);
                const float gg = tanhf_(dg[2][r]);
                const float og = sigmoidf_(dg[3][r]);
                const float cv = fg * c_reg[r][s] + ig * gg;
                c_reg[r][s] = cv;
                if (slot < E)
                    h_s[(r + 8 * lg) * 32 + slot] = og * tanhf_(cv);
            }
        }
        __syncthreads();
    }

    // Emit h_n (from LDS) and c_n (straight from registers).
    for (int e = lane; e < 16 * E; e += 32) {
        const int row = e / E, j = e % E;
        hN[(size_t)(b0 + row) * E + j] = h_s[row * 32 + j];
    }
#pragma unroll
    for (int s = 0; s < 2; ++s) {
        const int slot = ln + 16 * s;
        if (slot < E) {
#pragma unroll
            for (int r = 0; r < 8; ++r)
                cN[(size_t)(b0 + r + 8 * lg) * E + slot] = c_reg[r][s];
        }
    }
}

// ---------------------------------------------------------------------------
// Kernel 2: q = h_n @ Wq^T + bq ; k,v from c_n. One thread per (b, out-row).
// ---------------------------------------------------------------------------
__global__ void qkv_kernel(const float* __restrict__ hN, const float* __restrict__ cN,
                           const float* __restrict__ in_w, const float* __restrict__ in_b,
                           float* __restrict__ qB, float* __restrict__ kB,
                           float* __restrict__ vB)
{
    int idx = blockIdx.x * blockDim.x + threadIdx.x;
    if (idx >= Bn * 3 * E) return;
    const int b = idx / (3 * E);
    const int m = idx % (3 * E);
    const float* src = (m < E) ? &hN[(size_t)b * E] : &cN[(size_t)b * E];
    const float* w = &in_w[m * E];
    float acc = in_b[m];
#pragma unroll
    for (int k = 0; k < E; ++k) acc += src[k] * w[k];
    float* dst = (m < E) ? qB : (m < 2 * E) ? kB : vB;
    dst[(size_t)b * E + (m % E)] = acc;
}

// ---------------------------------------------------------------------------
// Kernel 3: flash attention over the batch axis. One wave = (head, 16 queries).
// Streams 256 key tiles; WMMA Q*K^T (K dim = HD padded to 32); online softmax;
// P*V accumulated in VALU from LDS (N=16, HD=10 per tile).
// ---------------------------------------------------------------------------
__global__ __launch_bounds__(32)
void attn_kernel(const float* __restrict__ qB, const float* __restrict__ kB,
                 const float* __restrict__ vB, float* __restrict__ ctxB)
{
    __shared__ float s_s[16 * 16];
    __shared__ float p_s[16 * 16];
    __shared__ float v_s[16 * HD];
    __shared__ float alpha_s[16];
    __shared__ float l_s[16];

    const int lane = threadIdx.x;
    const int lg = lane >> 4, ln = lane & 15;
    const int h  = blockIdx.x % NH;
    const int q0 = (blockIdx.x / NH) * 16;
    const float scale = rsqrtf((float)HD);

    // Q tile as A fragment, pre-scaled by 1/sqrt(HD).
    v16bf aq;
#pragma unroll
    for (int i = 0; i < 16; ++i) {
        const int kk = i + 16 * lg;
        float qv = (kk < HD) ? qB[(size_t)(q0 + ln) * E + h * HD + kk] * scale : 0.0f;
        aq[i] = (__bf16)qv;
    }

    float m_row = -INFINITY, l_row = 0.0f;   // valid on lanes 0..15 (row = ln)
    float ctx[5];
#pragma unroll
    for (int i2 = 0; i2 < 5; ++i2) ctx[i2] = 0.0f;
    const int e0 = lane * 5;                 // 160 = 16 rows x 10 dims

    for (int k0 = 0; k0 < Bn; k0 += 16) {
        // K^T tile as B fragment: N = key, K = head dim (10 of 32).
        v16bf bk;
#pragma unroll
        for (int i = 0; i < 16; ++i) {
            const int kk = i + 16 * lg;
            float kv = (kk < HD) ? kB[(size_t)(k0 + ln) * E + h * HD + kk] : 0.0f;
            bk[i] = (__bf16)kv;
        }
        v8f cz = {};
        v8f s = __builtin_amdgcn_wmma_f32_16x16x32_bf16(
            false, aq, false, bk, (short)0, cz, false, false);
#pragma unroll
        for (int r = 0; r < 8; ++r)
            s_s[(r + 8 * lg) * 16 + ln] = s[r];

        // Stage V tile (16 keys x HD).
        for (int e = lane; e < 16 * HD; e += 32)
            v_s[e] = vB[(size_t)(k0 + e / HD) * E + h * HD + (e % HD)];
        __syncthreads();

        if (lane < 16) {   // one lane per query row: online softmax stats
            float tmax = s_s[ln * 16];
#pragma unroll
            for (int n = 1; n < 16; ++n) tmax = fmaxf(tmax, s_s[ln * 16 + n]);
            const float mnew  = fmaxf(m_row, tmax);
            const float alpha = __expf(m_row - mnew);
            float psum = 0.0f;
#pragma unroll
            for (int n = 0; n < 16; ++n) {
                const float p = __expf(s_s[ln * 16 + n] - mnew);
                p_s[ln * 16 + n] = p;
                psum += p;
            }
            l_row = l_row * alpha + psum;
            m_row = mnew;
            alpha_s[ln] = alpha;
        }
        __syncthreads();

        // ctx = ctx*alpha + P*V   (5 (row,d) elements per lane)
#pragma unroll
        for (int i2 = 0; i2 < 5; ++i2) {
            const int row = (e0 + i2) / HD, d = (e0 + i2) % HD;
            float acc = ctx[i2] * alpha_s[row];
#pragma unroll
            for (int n = 0; n < 16; ++n)
                acc += p_s[row * 16 + n] * v_s[n * HD + d];
            ctx[i2] = acc;
        }
        __syncthreads();
    }

    if (lane < 16) l_s[ln] = l_row;
    __syncthreads();
#pragma unroll
    for (int i2 = 0; i2 < 5; ++i2) {
        const int row = (e0 + i2) / HD, d = (e0 + i2) % HD;
        ctxB[(size_t)(q0 + row) * E + h * HD + d] = ctx[i2] / l_s[row];
    }
}

// ---------------------------------------------------------------------------
// Kernel 4: out_proj + residual + LN + MLP(GELU exact) + LN + head. One lane
// per batch row; all 30x30 weights staged in LDS.
// ---------------------------------------------------------------------------
__global__ __launch_bounds__(32)
void tail_kernel(const float* __restrict__ ctxB, const float* __restrict__ hN,
                 const float* __restrict__ op_w, const float* __restrict__ op_b,
                 const float* __restrict__ f1w, const float* __restrict__ f1b,
                 const float* __restrict__ f2w, const float* __restrict__ f2b,
                 const float* __restrict__ lng, const float* __restrict__ lnb,
                 const float* __restrict__ ow, const float* __restrict__ ob,
                 float* __restrict__ out)
{
    __shared__ float opw_s[E * E], f1w_s[E * E], f2w_s[E * E], ow_s[3 * E];
    __shared__ float opb_s[E], f1b_s[E], f2b_s[E], lng_s[E], lnb_s[E], ob_s[3];

    const int lane = threadIdx.x;
    for (int e = lane; e < E * E; e += 32) {
        opw_s[e] = op_w[e]; f1w_s[e] = f1w[e]; f2w_s[e] = f2w[e];
    }
    for (int e = lane; e < 3 * E; e += 32) ow_s[e] = ow[e];
    if (lane < E) {
        opb_s[lane] = op_b[lane]; f1b_s[lane] = f1b[lane]; f2b_s[lane] = f2b[lane];
        lng_s[lane] = lng[lane];  lnb_s[lane] = lnb[lane];
    }
    if (lane < 3) ob_s[lane] = ob[lane];
    __syncthreads();

    const int b = blockIdx.x * 32 + lane;
    float cx[E], xa[E], x1v[E], yv[E];
#pragma unroll
    for (int k = 0; k < E; ++k) cx[k] = ctxB[(size_t)b * E + k];

    // attn_out + h_n
#pragma unroll
    for (int n = 0; n < E; ++n) {
        float acc = opb_s[n];
#pragma unroll
        for (int k = 0; k < E; ++k) acc += cx[k] * opw_s[n * E + k];
        xa[n] = acc + hN[(size_t)b * E + n];
    }
    // LayerNorm 1
    float mu = 0.0f;
#pragma unroll
    for (int n = 0; n < E; ++n) mu += xa[n];
    mu *= (1.0f / E);
    float var = 0.0f;
#pragma unroll
    for (int n = 0; n < E; ++n) { float d = xa[n] - mu; var += d * d; }
    var *= (1.0f / E);
    const float rs1 = rsqrtf(var + EPSI);
#pragma unroll
    for (int n = 0; n < E; ++n) x1v[n] = (xa[n] - mu) * rs1 * lng_s[n] + lnb_s[n];

    // fc1 + exact GELU
#pragma unroll
    for (int n = 0; n < E; ++n) {
        float acc = f1b_s[n];
#pragma unroll
        for (int k = 0; k < E; ++k) acc += x1v[k] * f1w_s[n * E + k];
        yv[n] = 0.5f * acc * (1.0f + erff(acc * 0.70710678118654752f));
    }
    // fc2 + residual
#pragma unroll
    for (int n = 0; n < E; ++n) {
        float acc = f2b_s[n];
#pragma unroll
        for (int k = 0; k < E; ++k) acc += yv[k] * f2w_s[n * E + k];
        xa[n] = x1v[n] + acc;
    }
    // LayerNorm 2
    mu = 0.0f;
#pragma unroll
    for (int n = 0; n < E; ++n) mu += xa[n];
    mu *= (1.0f / E);
    var = 0.0f;
#pragma unroll
    for (int n = 0; n < E; ++n) { float d = xa[n] - mu; var += d * d; }
    var *= (1.0f / E);
    const float rs2 = rsqrtf(var + EPSI);
#pragma unroll
    for (int n = 0; n < E; ++n) yv[n] = (xa[n] - mu) * rs2 * lng_s[n] + lnb_s[n];

    // head: [B,3]
#pragma unroll
    for (int m = 0; m < 3; ++m) {
        float acc = ob_s[m];
#pragma unroll
        for (int k = 0; k < E; ++k) acc += yv[k] * ow_s[m * E + k];
        out[(size_t)b * 3 + m] = acc;
    }
}

// ---------------------------------------------------------------------------
extern "C" void kernel_launch(void* const* d_in, const int* in_sizes, int n_in,
                              void* d_out, int out_size, void* d_ws, size_t ws_size,
                              hipStream_t stream)
{
    const float* input     = (const float*)d_in[0];
    const float* w_ih      = (const float*)d_in[1];
    const float* w_hh      = (const float*)d_in[2];
    const float* b_ih      = (const float*)d_in[3];
    const float* b_hh      = (const float*)d_in[4];
    const float* in_proj_w = (const float*)d_in[5];
    const float* in_proj_b = (const float*)d_in[6];
    const float* out_proj_w= (const float*)d_in[7];
    const float* out_proj_b= (const float*)d_in[8];
    const float* fc1_w     = (const float*)d_in[9];
    const float* fc1_b     = (const float*)d_in[10];
    const float* fc2_w     = (const float*)d_in[11];
    const float* fc2_b     = (const float*)d_in[12];
    const float* ln_g      = (const float*)d_in[13];
    const float* ln_b      = (const float*)d_in[14];
    const float* out_w     = (const float*)d_in[15];
    const float* out_b     = (const float*)d_in[16];
    float* out = (float*)d_out;

    float* ws   = (float*)d_ws;
    float* hN   = ws;
    float* cN   = hN + (size_t)Bn * E;
    float* qB   = cN + (size_t)Bn * E;
    float* kB   = qB + (size_t)Bn * E;
    float* vB   = kB + (size_t)Bn * E;
    float* ctxB = vB + (size_t)Bn * E;

    lstm_kernel<<<Bn / 16, 32, 0, stream>>>(input, w_ih, w_hh, b_ih, b_hh, hN, cN);

    const int qkvN = Bn * 3 * E;
    qkv_kernel<<<(qkvN + 255) / 256, 256, 0, stream>>>(hN, cN, in_proj_w, in_proj_b,
                                                       qB, kB, vB);

    attn_kernel<<<NH * (Bn / 16), 32, 0, stream>>>(qB, kB, vB, ctxB);

    tail_kernel<<<Bn / 32, 32, 0, stream>>>(ctxB, hN, out_proj_w, out_proj_b,
                                            fc1_w, fc1_b, fc2_w, fc2_b,
                                            ln_g, ln_b, out_w, out_b, out);
}
